// Caser_17497696764617
// MI455X (gfx1250) — matrix operand order repaired and checked by reading, not compiled, not run on hardware
//
#include <hip/hip_runtime.h>

typedef __bf16 v8bf  __attribute__((ext_vector_type(8)));
typedef __bf16 v16bf __attribute__((ext_vector_type(16)));
typedef float  v8f   __attribute__((ext_vector_type(8)));

union V16U { v16bf v; unsigned short s[16]; };

__device__ __forceinline__ unsigned short f2bf(float f) {
  unsigned u = __float_as_uint(f);
  u += 0x7fffu + ((u >> 16) & 1u);
  return (unsigned short)(u >> 16);
}

#define WMMA_BF16(a, b, c) \
  __builtin_amdgcn_wmma_f32_16x16x32_bf16(false, (a), false, (b), (short)0, (c), false, false)

// ---------------- problem constants ----------------
static constexpr int BB   = 1024;
static constexpr int LL   = 50;
static constexpr int DD   = 128;
static constexpr int NHH  = 16;
static constexpr int NVV  = 4;
static constexpr int NIT  = 100000;
static constexpr int FEAT = NVV * DD + NHH * LL;   // 1312 = 41*32
static constexpr int WH_ELEMS = NHH * DD * (LL * (LL + 1) / 2);  // 2,611,200

// ws layout (bytes), 256-aligned
static constexpr size_t WS_WH   = 0;
static constexpr size_t WS_FEAT = WS_WH   + (size_t)WH_ELEMS * 2;       // 5,222,400
static constexpr size_t WS_ZU   = WS_FEAT + (size_t)BB * FEAT * 2;      // +2,686,976
static constexpr size_t WS_SO   = WS_ZU   + (size_t)BB * 256 * 2;       // +524,288

struct WPtrs { const float* p[50]; };

// ---------------- weight conversion ----------------
__global__ __launch_bounds__(256) void convert_wh_concat(const float* __restrict__ src,
                                                         unsigned short* __restrict__ dst, int n) {
  for (int i = blockIdx.x * blockDim.x + threadIdx.x; i < n; i += gridDim.x * blockDim.x)
    dst[i] = f2bf(src[i]);
}

__global__ __launch_bounds__(256) void convert_wh_multi(WPtrs wp, unsigned short* __restrict__ dst) {
  int k = blockIdx.x;                       // 0..49
  int cnt = NHH * DD * (k + 1);
  int base = NHH * DD * (k * (k + 1) / 2);
  const float* s = wp.p[k];
  for (int e = threadIdx.x; e < cnt; e += blockDim.x) dst[base + e] = f2bf(s[e]);
}

// ---- one conv size k, NT N-tiles (statically unrolled accumulators) ----
template <int NT>
__device__ __forceinline__ void conv_run(
    const int k, const unsigned short* __restrict__ wh,
    const unsigned short* xs,                  // LDS bf16 x[b]
    const float* __restrict__ conv_h_bs,
    unsigned short* __restrict__ feat_b,       // &feat[b*FEAT]
    const int l15, const int hs) {
  const int nt     = LL - k;
  const int nchunk = (k + 1) * (DD / 32);
  const long wbase = (long)(NHH * DD) * (k * (k + 1) / 2);
  const int  rs    = (k + 1) * DD;             // row stride (halves) in w_k

  int tread[NT];
  #pragma unroll
  for (int t = 0; t < NT; ++t) {
    int tc = t * 16 + l15;
    tread[t] = tc < nt ? tc : nt - 1;          // clamp pad columns
  }
  v8f acc[NT];
  #pragma unroll
  for (int t = 0; t < NT; ++t) acc[t] = {0.f, 0.f, 0.f, 0.f, 0.f, 0.f, 0.f, 0.f};

  const unsigned short* pw = wh + wbase + (long)l15 * rs + hs * 16;
  V16U acur, anext;
  #pragma unroll
  for (int e = 0; e < 8; ++e) { acur.s[e] = pw[e]; acur.s[8 + e] = pw[8 + e]; }

  for (int c = 0; c < nchunk; ++c) {
    if (c + 1 < nchunk) {                      // prefetch next A fragment
      const unsigned short* pn = pw + (c + 1) * 32;
      #pragma unroll
      for (int e = 0; e < 8; ++e) { anext.s[e] = pn[e]; anext.s[8 + e] = pn[8 + e]; }
    }
    const int i     = c >> 2;
    const int dbase = ((c & 3) << 5) + (hs << 4);
    #pragma unroll
    for (int t = 0; t < NT; ++t) {             // A fragment reused across N-tiles
      const v16bf bv = *(const v16bf*)(&xs[(tread[t] + i) * DD + dbase]);
      acc[t] = WMMA_BF16(acur.v, bv, acc[t]);
    }
    acur = anext;
  }

  float hmax[8];
  #pragma unroll
  for (int r = 0; r < 8; ++r) hmax[r] = 0.f;
  #pragma unroll
  for (int t = 0; t < NT; ++t) {
    int tc = t * 16 + l15;
    #pragma unroll
    for (int r = 0; r < 8; ++r) {
      float v = acc[t][r] + conv_h_bs[k * NHH + hs * 8 + r];
      v = fmaxf(v, 0.f);
      if (tc >= nt) v = 0.f;
      hmax[r] = fmaxf(hmax[r], v);
    }
  }
  #pragma unroll
  for (int r = 0; r < 8; ++r)                  // max over the 16 N-columns
    for (int m = 1; m < 16; m <<= 1)
      hmax[r] = fmaxf(hmax[r], __shfl_xor(hmax[r], m, 32));
  if (l15 == 0) {
    #pragma unroll
    for (int r = 0; r < 8; ++r)
      feat_b[NVV * DD + k * NHH + hs * 8 + r] = f2bf(hmax[r]);
  }
}

// ---------------- feature kernel: async gather + out_v + h-convs (WMMA) ----------------
__global__ __launch_bounds__(256) void feature_kernel(
    const int* __restrict__ seq, const float* __restrict__ item_emb,
    const float* __restrict__ conv_v_w, const float* __restrict__ conv_v_b,
    const unsigned short* __restrict__ wh, const float* __restrict__ conv_h_bs,
    unsigned short* __restrict__ feat) {
  __shared__ float          xf[LL * DD];    // f32 staging, 25.6 KB (async DMA target)
  __shared__ unsigned short xs[LL * DD];    // x[b] in bf16, 12.8 KB
  __shared__ int sidx[LL];
  const int b    = blockIdx.x;
  const int tid  = threadIdx.x;
  const int lane = tid & 31;
  const int wv   = __builtin_amdgcn_readfirstlane(tid >> 5);  // scalar wave id
  const int l15  = lane & 15;
  const int hs   = lane >> 4;               // 0 or 1

  if (tid < LL) sidx[tid] = seq[b * LL + tid];
  __syncthreads();

  // Async DMA gather: per-lane global -> LDS, 16B granules, tracked by ASYNCcnt.
  for (int q = tid; q < LL * DD / 4; q += 256) {     // 1600 float4 granules
    int l  = q >> 5;                                 // 32 granules per row
    int c4 = q & 31;
    const float* gp = item_emb + (long)sidx[l] * DD + c4 * 4;
    unsigned ldsb = (unsigned)(uintptr_t)(&xf[q * 4]);  // flat LDS addr low 32b = LDS offset
    asm volatile("global_load_async_to_lds_b128 %0, %1, off"
                 :: "v"(ldsb), "v"(gp) : "memory");
  }
  asm volatile("s_wait_asynccnt 0x0" ::: "memory");
  __syncthreads();
  for (int e = tid; e < LL * DD; e += 256) xs[e] = f2bf(xf[e]);
  __syncthreads();

  // out_v: (4 x 128), 2 outputs/thread
  for (int o = tid; o < NVV * DD; o += 256) {
    int v = o >> 7, d = o & 127;
    float acc = conv_v_b[v];
    #pragma unroll 10
    for (int l = 0; l < LL; ++l)
      acc += conv_v_w[v * LL + l] * xf[l * DD + d];
    feat[(long)b * FEAT + o] = f2bf(acc);
  }

  // out_h: wave wv handles k = wv, wv+8, ...  (k is scalar -> uniform control flow)
  unsigned short* feat_b = feat + (long)b * FEAT;
  for (int k = wv; k < LL; k += 8) {
    const int ntiles = (LL - k + 15) >> 4;
    if (ntiles == 4)      conv_run<4>(k, wh, xs, conv_h_bs, feat_b, l15, hs);
    else if (ntiles == 3) conv_run<3>(k, wh, xs, conv_h_bs, feat_b, l15, hs);
    else if (ntiles == 2) conv_run<2>(k, wh, xs, conv_h_bs, feat_b, l15, hs);
    else                  conv_run<1>(k, wh, xs, conv_h_bs, feat_b, l15, hs);
  }
}

// ---------------- user gather into zu[:,128:256] ----------------
__global__ __launch_bounds__(256) void gather_user(const int* __restrict__ user_id,
                                                   const float* __restrict__ uet,
                                                   unsigned short* __restrict__ zu) {
  int idx = blockIdx.x * blockDim.x + threadIdx.x;   // 0..131071
  int b = idx >> 7, d = idx & 127;
  zu[b * 256 + 128 + d] = f2bf(uet[(long)user_id[b] * DD + d]);
}

// ---------------- generic fc GEMM: (1024 x K)bf16 @ (128 x K)f32^T, bias+relu, bf16 out ----------
__global__ __launch_bounds__(256) void fc_gemm_kernel(
    const unsigned short* __restrict__ Abf, int lda,
    const float* __restrict__ Wf, const float* __restrict__ bias, int K,
    unsigned short* __restrict__ Out, int ldo) {
  const int lane = threadIdx.x & 31;
  const int wv   = __builtin_amdgcn_readfirstlane(threadIdx.x >> 5);
  const int l15 = lane & 15, hs = lane >> 4;
  const int mbase = blockIdx.x * 16;              // 64 blocks cover M=1024
  const int n = wv * 16 + l15;                    // 8 waves cover N=128
  v8f acc = {0.f, 0.f, 0.f, 0.f, 0.f, 0.f, 0.f, 0.f};
  const int nchunk = K >> 5;
  for (int c = 0; c < nchunk; ++c) {
    const unsigned short* pa = Abf + (long)(mbase + l15) * lda + c * 32 + hs * 16;
    V16U au;
    #pragma unroll
    for (int e = 0; e < 8; ++e) { au.s[e] = pa[e]; au.s[8 + e] = pa[8 + e]; }
    const float* pb = Wf + (long)n * K + c * 32 + hs * 16;
    V16U bu;
    #pragma unroll
    for (int e = 0; e < 16; ++e) bu.s[e] = f2bf(pb[e]);
    acc = WMMA_BF16(au.v, bu.v, acc);
  }
  const float bv = bias[n];
  #pragma unroll
  for (int r = 0; r < 8; ++r) {
    int m = mbase + hs * 8 + r;
    Out[(long)m * ldo + n] = f2bf(fmaxf(acc[r] + bv, 0.f));
  }
}

// ---------------- logits: (1024 x 128)bf16 @ item_emb^T -> (1024 x 100000)f32 ----------------
__global__ __launch_bounds__(256) void logits_kernel(
    const unsigned short* __restrict__ seqout, const float* __restrict__ item_emb,
    float* __restrict__ out) {
  __shared__ unsigned short it_s[16 * DD];        // bf16 item tile, 4 KB, shared by 8 waves
  const int tid = threadIdx.x;
  const int lane = tid & 31;
  const int wv   = __builtin_amdgcn_readfirstlane(tid >> 5);
  const int l15 = lane & 15, hs = lane >> 4;
  const int nbase = blockIdx.x * 16;              // 6250 blocks cover N=100000

  for (int e = tid; e < 16 * DD; e += 256) {      // convert tile once per block
    int r = e >> 7, d = e & 127;
    it_s[e] = f2bf(item_emb[(long)(nbase + r) * DD + d]);
  }
  __syncthreads();

  v16bf bfr[4];
  #pragma unroll
  for (int c = 0; c < 4; ++c)
    bfr[c] = *(const v16bf*)(&it_s[l15 * DD + c * 32 + hs * 16]);

  V16U a[4], an[4];
  #pragma unroll
  for (int c = 0; c < 4; ++c) {                   // preload first M-tile's A fragments
    const unsigned short* pa = seqout + (long)(wv * 16 + l15) * DD + c * 32 + hs * 16;
    #pragma unroll
    for (int e = 0; e < 8; ++e) { a[c].s[e] = pa[e]; a[c].s[8 + e] = pa[8 + e]; }
  }

  for (int mt = wv; mt < BB / 16; mt += 8) {      // 8 M-tiles per wave (scalar loop)
    if (mt + 8 < BB / 16) {                       // prefetch next M-tile's A
      #pragma unroll
      for (int c = 0; c < 4; ++c) {
        const unsigned short* pa =
            seqout + (long)((mt + 8) * 16 + l15) * DD + c * 32 + hs * 16;
        #pragma unroll
        for (int e = 0; e < 8; ++e) { an[c].s[e] = pa[e]; an[c].s[8 + e] = pa[8 + e]; }
      }
    }
    v8f acc = {0.f, 0.f, 0.f, 0.f, 0.f, 0.f, 0.f, 0.f};
    #pragma unroll
    for (int c = 0; c < 4; ++c) acc = WMMA_BF16(a[c].v, bfr[c], acc);
    #pragma unroll
    for (int r = 0; r < 8; ++r) {
      int m = mt * 16 + hs * 8 + r;
      out[(long)m * NIT + nbase + l15] = acc[r];
    }
    #pragma unroll
    for (int c = 0; c < 4; ++c) a[c] = an[c];
  }
}

// ---------------- host launcher ----------------
extern "C" void kernel_launch(void* const* d_in, const int* in_sizes, int n_in,
                              void* d_out, int out_size, void* d_ws, size_t ws_size,
                              hipStream_t stream) {
  const int*   seq      = (const int*)d_in[0];
  const int*   user_id  = (const int*)d_in[1];
  const float* item_emb = (const float*)d_in[2];
  const float* user_emb = (const float*)d_in[3];
  const float* conv_v_w = (const float*)d_in[4];
  const float* conv_v_b = (const float*)d_in[5];

  unsigned short* wh     = (unsigned short*)((char*)d_ws + WS_WH);
  unsigned short* feat   = (unsigned short*)((char*)d_ws + WS_FEAT);
  unsigned short* zu     = (unsigned short*)((char*)d_ws + WS_ZU);
  unsigned short* seqout = (unsigned short*)((char*)d_ws + WS_SO);

  int base;
  if (n_in == 12) {
    // conv_h_ws flattened/concatenated into one buffer
    convert_wh_concat<<<1024, 256, 0, stream>>>((const float*)d_in[6], wh, WH_ELEMS);
    base = 7;
  } else {
    // 50 separate tuple entries
    WPtrs wp;
    for (int k = 0; k < 50; ++k) wp.p[k] = (const float*)d_in[6 + k];
    convert_wh_multi<<<50, 256, 0, stream>>>(wp, wh);
    base = 56;
  }
  const float* conv_h_bs = (const float*)d_in[base];
  const float* fc1_w     = (const float*)d_in[base + 1];
  const float* fc1_b     = (const float*)d_in[base + 2];
  const float* fc2_w     = (const float*)d_in[base + 3];
  const float* fc2_b     = (const float*)d_in[base + 4];

  feature_kernel<<<BB, 256, 0, stream>>>(seq, item_emb, conv_v_w, conv_v_b, wh, conv_h_bs, feat);
  gather_user<<<(BB * DD) / 256, 256, 0, stream>>>(user_id, user_emb, zu);
  fc_gemm_kernel<<<BB / 16, 256, 0, stream>>>(feat, FEAT, fc1_w, fc1_b, FEAT, zu, 256);
  fc_gemm_kernel<<<BB / 16, 256, 0, stream>>>(zu, 256, fc2_w, fc2_b, 256, seqout, 128);
  logits_kernel<<<NIT / 16, 256, 0, stream>>>(seqout, item_emb, (float*)d_out);
}
